// CustomLSTM_16999480558292
// MI455X (gfx1250) — compile-verified, matrix operand order checked by reading, not run on hardware
//
#include <hip/hip_runtime.h>
#include <hip/hip_bf16.h>
#include <math.h>

// ---------------- problem dims (fixed by reference) ----------------
#define TT 512
#define BB 64
#define II 1024
#define HH 1024
#define G4 4096   // 4*H
#define LL 2

typedef __bf16 bf16;
typedef __attribute__((ext_vector_type(16))) __bf16 v16bf;
typedef __attribute__((ext_vector_type(8)))  __bf16 v8bf;
typedef __attribute__((ext_vector_type(8)))  float  v8f;

static __device__ __forceinline__ bf16 f2bf(float f) {
  unsigned u = __builtin_bit_cast(unsigned, f);
  unsigned r = u + 0x7FFFu + ((u >> 16) & 1u);   // round-to-nearest-even
  unsigned short s = (unsigned short)(r >> 16);
  return __builtin_bit_cast(bf16, s);
}
static __device__ __forceinline__ float sigmoidf_(float x) {
  return 1.0f / (1.0f + __expf(-x));
}

// async copy of 16 bytes global -> LDS (ASYNCcnt-tracked).
// ldsOff is a byte offset relative to the workgroup LDS base (we only use
// dynamic shared memory, which starts at offset 0).
static __device__ __forceinline__ void async_g2l_b128(unsigned ldsOff, const void* g) {
  asm volatile("global_load_async_to_lds_b128 %0, %1, off"
               :: "v"(ldsOff), "v"((unsigned long long)(uintptr_t)g)
               : "memory");
}
static __device__ __forceinline__ void wait_async0() {
  asm volatile("s_wait_asynccnt 0x0" ::: "memory");
}

// Read a 16x32 bf16 WMMA operand fragment out of an LDS tile stored
// row-major [64 rows][64 k] (128 bytes per row).
// ISA 16-bit operand layout (A and B use the same per-lane mapping):
//   lanes 0-15 : elems 0-7 -> K=+0..7,  elems 8-15 -> K=+16..23
//   lanes 16-31: elems 0-7 -> K=+8..15, elems 8-15 -> K=+24..31
static __device__ __forceinline__ v16bf lds_frag(const char* smem, unsigned byteOff) {
  const v8bf* p = (const v8bf*)(smem + byteOff);
  v8bf lo = p[0];      // +0  bytes
  v8bf hi = p[2];      // +32 bytes
  return __builtin_shufflevector(lo, hi, 0,1,2,3,4,5,6,7,8,9,10,11,12,13,14,15);
}

// C[M,N] = A[M,K](bf16) * W[N,K]^T(bf16) + (Cin ? Cin : 0) + (bias ? bias : 0)
// Block = 4 waves; block tile 64M x 64N; wave w owns M-rows [64*by+16w, +16)
// x all 64 N columns. K staged through LDS 64 at a time (two 32-deep WMMA
// sub-steps per stage), double buffered, fed by global_load_async_to_lds_b128.
// M and N must be multiples of 64; K a multiple of 64.
__global__ void __launch_bounds__(128)
gemm_bf16_wmma(const bf16* __restrict__ A, const bf16* __restrict__ W,
               const float* __restrict__ Cin, const float* __restrict__ bias,
               float* __restrict__ C, int M, int N, int K) {
  (void)M;
  extern __shared__ char smem[];        // 2 * (8KB A-tile + 8KB W-tile)
  const int tid   = threadIdx.x;
  const int lane  = tid & 31;
  const int wave  = tid >> 5;
  const int mBase = blockIdx.y * 64;
  const int nBase = blockIdx.x * 64;
  const int r     = lane & 15;
  const int half  = lane >> 4;

  // ---- stage one 64-wide K slab of A(64 rows) and W(64 rows) into LDS ----
  auto stage = [&](int buf, int k0) {
    const unsigned base = (unsigned)buf * 16384u;
#pragma unroll
    for (int i = 0; i < 4; ++i) {
      const int c   = tid + i * 128;       // chunk id 0..511 (16B each)
      const int row = c >> 3;              // 0..63
      const int kc  = (c & 7) * 8;         // element offset within slab
      async_g2l_b128(base + (unsigned)c * 16u,
                     A + (size_t)(mBase + row) * K + k0 + kc);
      async_g2l_b128(base + 8192u + (unsigned)c * 16u,
                     W + (size_t)(nBase + row) * K + k0 + kc);
    }
  };

  v8f acc[4];
  if (Cin) {
#pragma unroll
    for (int j = 0; j < 4; ++j)
#pragma unroll
      for (int rr = 0; rr < 8; ++rr)
        acc[j][rr] = Cin[(size_t)(mBase + 16 * wave + rr + 8 * half) * N + nBase + j * 16 + r];
  } else {
#pragma unroll
    for (int j = 0; j < 4; ++j) acc[j] = (v8f){};
  }

  stage(0, 0);
  int buf = 0;
  for (int k0 = 0; k0 < K; k0 += 64) {
    wait_async0();        // this wave's copies for `buf` (and older) complete
    __syncthreads();      // => every wave's copies for `buf` complete
    if (k0 + 64 < K) stage(buf ^ 1, k0 + 64);   // overlap next slab with compute

    const unsigned bA = (unsigned)buf * 16384u;
#pragma unroll
    for (int s = 0; s < 2; ++s) {                // two 32-deep sub-steps
      const unsigned sOff = (unsigned)s * 64u + (unsigned)half * 16u;
      const unsigned aOff = bA + (unsigned)(16 * wave + r) * 128u + sOff;
      const unsigned w0   = bA + 16384u / 2u + (unsigned)r * 128u + sOff;
      // issue all fragment loads first so they clause into one ds burst
      v16bf a  = lds_frag(smem, aOff);
      v16bf b0 = lds_frag(smem, w0);
      v16bf b1 = lds_frag(smem, w0 + 16u * 128u);
      v16bf b2 = lds_frag(smem, w0 + 32u * 128u);
      v16bf b3 = lds_frag(smem, w0 + 48u * 128u);
      acc[0] = __builtin_amdgcn_wmma_f32_16x16x32_bf16(false, a, false, b0,
                                                       (short)0, acc[0], false, false);
      acc[1] = __builtin_amdgcn_wmma_f32_16x16x32_bf16(false, a, false, b1,
                                                       (short)0, acc[1], false, false);
      acc[2] = __builtin_amdgcn_wmma_f32_16x16x32_bf16(false, a, false, b2,
                                                       (short)0, acc[2], false, false);
      acc[3] = __builtin_amdgcn_wmma_f32_16x16x32_bf16(false, a, false, b3,
                                                       (short)0, acc[3], false, false);
    }
    buf ^= 1;
  }

  // D layout: VGPR rr, lane L -> M = rr + 8*(L>=16), N = L&15
#pragma unroll
  for (int j = 0; j < 4; ++j) {
    const int n = nBase + j * 16 + r;
    const float bv = bias ? bias[n] : 0.0f;
#pragma unroll
    for (int rr = 0; rr < 8; ++rr)
      C[(size_t)(mBase + 16 * wave + rr + 8 * half) * N + n] = acc[j][rr] + bv;
  }
}

// elementwise LSTM cell update for one timestep
__global__ void __launch_bounds__(256)
lstm_cell(const float* __restrict__ gates,                  // [B,4H]
          float* __restrict__ h, float* __restrict__ c,     // [B,H]
          bf16* __restrict__ hbf,                           // [B,H]
          float* __restrict__ seq_f32,                      // [B,H] or null
          bf16* __restrict__ seq_bf16) {                    // [B,H] or null
  int idx = blockIdx.x * blockDim.x + threadIdx.x;
  if (idx >= BB * HH) return;
  int b = idx >> 10, j = idx & (HH - 1);
  const float* g = gates + (size_t)b * G4;
  float gi = g[j], gf = g[j + HH], gg = g[j + 2 * HH], go = g[j + 3 * HH];
  float cN = sigmoidf_(gf) * c[idx] + sigmoidf_(gi) * tanhf(gg);
  float hN = sigmoidf_(go) * tanhf(cN);
  c[idx] = cN;
  h[idx] = hN;
  bf16 hb = f2bf(hN);
  hbf[idx] = hb;
  if (seq_f32)  seq_f32[idx]  = hN;
  if (seq_bf16) seq_bf16[idx] = hb;
}

__global__ void __launch_bounds__(256)
zero_state(float* __restrict__ h, float* __restrict__ c, bf16* __restrict__ hbf, int n) {
  int i = blockIdx.x * blockDim.x + threadIdx.x;
  if (i < n) { h[i] = 0.0f; c[i] = 0.0f; hbf[i] = f2bf(0.0f); }
}

__global__ void __launch_bounds__(256)
cvt_f32_bf16(const float* __restrict__ src, bf16* __restrict__ dst, int n) {
  int i = blockIdx.x * blockDim.x + threadIdx.x;
  if (i < n) dst[i] = f2bf(src[i]);
}

__global__ void __launch_bounds__(256)
bias_sum(const float* __restrict__ a, const float* __restrict__ b,
         float* __restrict__ out, int n) {
  int i = blockIdx.x * blockDim.x + threadIdx.x;
  if (i < n) out[i] = a[i] + b[i];
}

__global__ void __launch_bounds__(256)
copy_tail(const float* __restrict__ h0, const float* __restrict__ h1,
          const float* __restrict__ c0, const float* __restrict__ c1,
          float* __restrict__ out) {
  int i = blockIdx.x * blockDim.x + threadIdx.x;
  const int n = BB * HH;
  if (i < n) {
    out[i]         = h0[i];
    out[n + i]     = h1[i];
    out[2 * n + i] = c0[i];
    out[3 * n + i] = c1[i];
  }
}

extern "C" void kernel_launch(void* const* d_in, const int* in_sizes, int n_in,
                              void* d_out, int out_size, void* d_ws, size_t ws_size,
                              hipStream_t stream) {
  (void)in_sizes; (void)n_in; (void)out_size; (void)ws_size;
  const float* x    = (const float*)d_in[0];  // [T,B,I]
  const float* w_ih = (const float*)d_in[1];  // [L,4H,I]
  const float* w_hh = (const float*)d_in[2];  // [L,4H,H]
  const float* b_ih = (const float*)d_in[3];  // [L,4H]
  const float* b_hh = (const float*)d_in[4];  // [L,4H]
  float* out = (float*)d_out;                 // seq [T,B,H] + h_n [2,B,H] + c_n [2,B,H]

  // ---- workspace carve-up ----
  char* ws = (char*)d_ws;
  size_t off = 0;
  auto carve = [&](size_t bytes) { char* p = ws + off; off += bytes; return p; };
  bf16*  xbf    = (bf16*)carve((size_t)TT * BB * II * sizeof(bf16));
  bf16*  seq1bf = (bf16*)carve((size_t)TT * BB * HH * sizeof(bf16));
  bf16*  wihbf  = (bf16*)carve((size_t)LL * G4 * II * sizeof(bf16));
  bf16*  whhbf  = (bf16*)carve((size_t)LL * G4 * HH * sizeof(bf16));
  float* xin    = (float*)carve((size_t)TT * BB * G4 * sizeof(float));
  float* gates  = (float*)carve((size_t)BB * G4 * sizeof(float));
  float* hbuf   = (float*)carve((size_t)LL * BB * HH * sizeof(float));
  float* cbuf   = (float*)carve((size_t)LL * BB * HH * sizeof(float));
  bf16*  hbf    = (bf16*)carve((size_t)BB * HH * sizeof(bf16));
  float* biasb  = (float*)carve((size_t)LL * G4 * sizeof(float));

  // ---- weight / input conversion to bf16 ----
  {
    int n = TT * BB * II;
    cvt_f32_bf16<<<(n + 255) / 256, 256, 0, stream>>>(x, xbf, n);
    n = LL * G4 * II;
    cvt_f32_bf16<<<(n + 255) / 256, 256, 0, stream>>>(w_ih, wihbf, n);
    n = LL * G4 * HH;
    cvt_f32_bf16<<<(n + 255) / 256, 256, 0, stream>>>(w_hh, whhbf, n);
    n = LL * G4;
    bias_sum<<<(n + 255) / 256, 256, 0, stream>>>(b_ih, b_hh, biasb, n);
  }

  const int nState = BB * HH;
  const size_t smemBytes = 32768;   // 2 buffers * (8KB A + 8KB W)
  for (int l = 0; l < LL; ++l) {
    zero_state<<<(nState + 255) / 256, 256, 0, stream>>>(
        hbuf + (size_t)l * nState, cbuf + (size_t)l * nState, hbf, nState);

    // xin[t,b,:] = seq[t,b,:] @ wi^T + (b_ih+b_hh)   -- one big GEMM
    {
      dim3 grid(G4 / 64, (TT * BB) / 64);   // (64, 512)
      gemm_bf16_wmma<<<grid, 128, smemBytes, stream>>>(
          (l == 0) ? xbf : seq1bf,
          wihbf + (size_t)l * G4 * II,
          nullptr, biasb + (size_t)l * G4,
          xin, TT * BB, G4, II);
    }

    // sequential scan
    for (int t = 0; t < TT; ++t) {
      dim3 grid(G4 / 64, 1);                 // M=64 -> one 64-row slab
      gemm_bf16_wmma<<<grid, 128, smemBytes, stream>>>(
          hbf, whhbf + (size_t)l * G4 * HH,
          xin + (size_t)t * BB * G4, nullptr,
          gates, BB, G4, HH);

      float* seqF = (l == LL - 1) ? (out + (size_t)t * BB * HH) : nullptr;
      bf16*  seqB = (l == 0)      ? (seq1bf + (size_t)t * BB * HH) : nullptr;
      lstm_cell<<<(nState + 255) / 256, 256, 0, stream>>>(
          gates, hbuf + (size_t)l * nState, cbuf + (size_t)l * nState,
          hbf, seqF, seqB);
    }
  }

  copy_tail<<<(nState + 255) / 256, 256, 0, stream>>>(
      hbuf, hbuf + nState, cbuf, cbuf + nState, out + (size_t)TT * BB * HH);
}
